// Mamba_4071628996980
// MI455X (gfx1250) — matrix operand order repaired and checked by reading, not compiled, not run on hardware
//
#include <hip/hip_runtime.h>
#include <hip/hip_bf16.h>

// ---------------- problem constants ----------------
#define BSZ     2
#define LSEQ    4096
#define DMODEL  1024
#define DIN     2048      // EXPAND * DMODEL
#define NSTATE  16
#define DTRANK  64
#define DCONV   4
#define NROWS   (BSZ * LSEQ)   // 8192 GEMM rows
#define NCHUNK  32
#define CLEN    128            // LSEQ / NCHUNK

#define LDS_STRIDE 48          // bf16 row stride: 96B, 16B aligned, conflict-skewed

// ---------------- WMMA vector types ----------------
typedef __attribute__((ext_vector_type(16))) __bf16 v16bf;
typedef __attribute__((ext_vector_type(8)))  float  v8f;

union FragBF {
    v16bf v;
    uint4 q[2];
};

__device__ __forceinline__ unsigned f2bf_pk(float a, float b) {
    union { float f; unsigned u; } x, y; x.f = a; y.f = b;
    unsigned ra = (x.u + 0x7FFFu + ((x.u >> 16) & 1u)) >> 16;   // RNE
    unsigned rb = (y.u + 0x7FFFu + ((y.u >> 16) & 1u)) & 0xFFFF0000u;
    return ra | rb;
}

__device__ __forceinline__ float sigmoidf_fast(float x) {
    return 1.0f / (1.0f + __expf(-x));
}

__device__ __forceinline__ float softplusf(float x) {
    return (x > 20.0f) ? x : log1pf(__expf(x));
}

// =====================================================================
// Generic GEMM: C[M,N] = A[M,K] * W[N,K]^T   (fp32 in/out, bf16 WMMA)
// M%128==0, N%128==0, K%32==0. grid=(N/128, M/128), block 256 (8 waves).
// Double-buffered LDS; next tile's global loads issued before the 8 WMMAs.
// =====================================================================
__launch_bounds__(256)
__global__ void gemm_bf16_wmma(const float* __restrict__ A, int lda,
                               const float* __restrict__ W, int ldw,
                               float* __restrict__ C, int ldc, int K) {
    __shared__ unsigned short As[2][128 * LDS_STRIDE];
    __shared__ unsigned short Bs[2][128 * LDS_STRIDE];

    const int tid    = threadIdx.x;
    const int lane   = tid & 31;
    const int wave   = tid >> 5;
    const int wr     = wave & 3;             // wave row (M dir), 4
    const int wc     = wave >> 2;            // wave col (N dir), 2
    const int half   = lane >> 4;
    const int lane16 = lane & 15;

    const size_t blockN = (size_t)blockIdx.x * 128;
    const size_t blockM = (size_t)blockIdx.y * 128;

    // per-thread staging coordinates: 4 x float4 per matrix per K-step
    int sm[4], sk[4];
#pragma unroll
    for (int r = 0; r < 4; ++r) {
        const int f = tid + 256 * r;          // 0..1023
        sm[r] = f >> 3;                       // row 0..127
        sk[r] = (f & 7) << 2;                 // k 0,4,...,28
    }

    const v8f vzero = {0.f,0.f,0.f,0.f,0.f,0.f,0.f,0.f};
    v8f acc[2][4];
#pragma unroll
    for (int mi = 0; mi < 2; ++mi)
#pragma unroll
        for (int ni = 0; ni < 4; ++ni) acc[mi][ni] = vzero;

    float4 ra[4], rw[4];

    // ---- prologue: load + stage tile k0 = 0 into buffer 0 ----
#pragma unroll
    for (int r = 0; r < 4; ++r) {
        ra[r] = *reinterpret_cast<const float4*>(A + (blockM + sm[r]) * (size_t)lda + sk[r]);
        rw[r] = *reinterpret_cast<const float4*>(W + (blockN + sm[r]) * (size_t)ldw + sk[r]);
    }
#pragma unroll
    for (int r = 0; r < 4; ++r) {
        *reinterpret_cast<uint2*>(&As[0][sm[r] * LDS_STRIDE + sk[r]]) =
            make_uint2(f2bf_pk(ra[r].x, ra[r].y), f2bf_pk(ra[r].z, ra[r].w));
        *reinterpret_cast<uint2*>(&Bs[0][sm[r] * LDS_STRIDE + sk[r]]) =
            make_uint2(f2bf_pk(rw[r].x, rw[r].y), f2bf_pk(rw[r].z, rw[r].w));
    }
    __syncthreads();

    for (int k0 = 0; k0 < K; k0 += 32) {
        const int  buf      = (k0 >> 5) & 1;
        const bool has_next = (k0 + 32) < K;

        // ---- issue next tile's global loads (no wait yet) ----
        if (has_next) {
#pragma unroll
            for (int r = 0; r < 4; ++r) {
                ra[r] = *reinterpret_cast<const float4*>(
                    A + (blockM + sm[r]) * (size_t)lda + k0 + 32 + sk[r]);
                rw[r] = *reinterpret_cast<const float4*>(
                    W + (blockN + sm[r]) * (size_t)ldw + k0 + 32 + sk[r]);
            }
        }

        // ---- 8 WMMAs per wave from current LDS buffer ----
#pragma unroll
        for (int mi = 0; mi < 2; ++mi) {
            FragBF af;
            // A 16x32 bf16 fragment (ISA 7.12.2): lanes 0-15 K0-7/16-23, lanes 16-31 K8-15/24-31
            const uint4* rowA = reinterpret_cast<const uint4*>(
                &As[buf][(wr * 32 + mi * 16 + lane16) * LDS_STRIDE]);
            af.q[0] = rowA[half];
            af.q[1] = rowA[2 + half];
#pragma unroll
            for (int ni = 0; ni < 4; ++ni) {
                FragBF bf;
                // B 32x16 fragment: col per lane; lanes 0-15 K0-7/16-23, lanes 16-31 K8-15/24-31
                const uint4* rowB = reinterpret_cast<const uint4*>(
                    &Bs[buf][(wc * 64 + ni * 16 + lane16) * LDS_STRIDE]);
                bf.q[0] = rowB[half];
                bf.q[1] = rowB[2 + half];
                acc[mi][ni] = __builtin_amdgcn_wmma_f32_16x16x32_bf16(
                    false, af.v, false, bf.v, (short)0, acc[mi][ni], false, false);
            }
        }

        // ---- convert + store next tile into the other buffer ----
        if (has_next) {
#pragma unroll
            for (int r = 0; r < 4; ++r) {
                *reinterpret_cast<uint2*>(&As[buf ^ 1][sm[r] * LDS_STRIDE + sk[r]]) =
                    make_uint2(f2bf_pk(ra[r].x, ra[r].y), f2bf_pk(ra[r].z, ra[r].w));
                *reinterpret_cast<uint2*>(&Bs[buf ^ 1][sm[r] * LDS_STRIDE + sk[r]]) =
                    make_uint2(f2bf_pk(rw[r].x, rw[r].y), f2bf_pk(rw[r].z, rw[r].w));
            }
        }
        __syncthreads();
    }

    // ---- writeback: C/D layout: VGPR j -> M = j + 8*half, N = lane16 ----
#pragma unroll
    for (int mi = 0; mi < 2; ++mi)
#pragma unroll
        for (int ni = 0; ni < 4; ++ni)
#pragma unroll
            for (int j = 0; j < 8; ++j) {
                const size_t m = blockM + wr * 32 + mi * 16 + half * 8 + j;
                const size_t n = blockN + wc * 64 + ni * 16 + lane16;
                C[m * (size_t)ldc + n] = acc[mi][ni][j];
            }
}

// =====================================================================
// Zero-pad x_proj_w [96,2048] -> [128,2048]
// =====================================================================
__launch_bounds__(256)
__global__ void pad_xproj_w(const float* __restrict__ w, float* __restrict__ wp) {
    const int idx = blockIdx.x * 256 + threadIdx.x;   // 128*2048
    const int row = idx >> 11;
    const int col = idx & 2047;
    wp[idx] = (row < (DTRANK + 2 * NSTATE)) ? w[row * DIN + col] : 0.0f;
}

// =====================================================================
// Depthwise causal conv(k=4) + bias + SiLU on u = xz[:, :, 0:2048]
// =====================================================================
__launch_bounds__(256)
__global__ void conv_silu_kernel(const float* __restrict__ xz,
                                 const float* __restrict__ cw,
                                 const float* __restrict__ cb,
                                 float* __restrict__ u_act) {
    const size_t idx = (size_t)blockIdx.x * 256 + threadIdx.x;  // B*L*DIN
    const int d  = (int)(idx % DIN);
    const size_t bl = idx / DIN;
    const int l  = (int)(bl % LSEQ);
    const size_t brow = bl - l;                                  // b*LSEQ
    float acc = cb[d];
#pragma unroll
    for (int j = 0; j < DCONV; ++j) {
        const int li = l - (DCONV - 1) + j;
        if (li >= 0)
            acc += cw[d * DCONV + j] * xz[(brow + li) * (size_t)(2 * DIN) + d];
    }
    u_act[idx] = acc * sigmoidf_fast(acc);
}

// =====================================================================
// Scan pass A: per (b,d,chunk) zero-init scan -> h_end[16], sum(delta)
// =====================================================================
__launch_bounds__(256)
__global__ void scan_passA(const float* __restrict__ dtraw, const float* __restrict__ dtb,
                           const float* __restrict__ u_act, const float* __restrict__ x_dbl,
                           const float* __restrict__ A_log,
                           float* __restrict__ SD, float* __restrict__ HE) {
    const int d = blockIdx.x * 256 + threadIdx.x;
    const int c = blockIdx.y, b = blockIdx.z;
    const float bias = dtb[d];
    float An[NSTATE], h[NSTATE];
#pragma unroll
    for (int n = 0; n < NSTATE; ++n) { An[n] = -__expf(A_log[d * NSTATE + n]); h[n] = 0.0f; }
    float sumd = 0.0f;
    const size_t base = (size_t)b * LSEQ + (size_t)c * CLEN;
    for (int i = 0; i < CLEN; ++i) {
        const size_t row = base + i;
        const float delta = softplusf(dtraw[row * DIN + d] + bias);
        const float du    = delta * u_act[row * DIN + d];
        sumd += delta;
        const float* bc = x_dbl + row * 128 + DTRANK;   // B_t[0..15]
#pragma unroll
        for (int n = 0; n < NSTATE; ++n)
            h[n] = h[n] * __expf(delta * An[n]) + du * bc[n];
    }
    const size_t cb2 = (size_t)b * NCHUNK + c;
    SD[cb2 * DIN + d] = sumd;
#pragma unroll
    for (int n = 0; n < NSTATE; ++n)
        HE[(cb2 * NSTATE + n) * DIN + d] = h[n];
}

// =====================================================================
// Scan pass B: 32-step chunk-carry recurrence per (b,d)
// =====================================================================
__launch_bounds__(256)
__global__ void scan_passB(const float* __restrict__ SD, const float* __restrict__ HE,
                           const float* __restrict__ A_log, float* __restrict__ HIN) {
    const int d = blockIdx.x * 256 + threadIdx.x;
    const int b = blockIdx.y;
    float An[NSTATE], carry[NSTATE];
#pragma unroll
    for (int n = 0; n < NSTATE; ++n) { An[n] = -__expf(A_log[d * NSTATE + n]); carry[n] = 0.0f; }
    for (int c = 0; c < NCHUNK; ++c) {
        const size_t cb2 = (size_t)b * NCHUNK + c;
        const float sd = SD[cb2 * DIN + d];
#pragma unroll
        for (int n = 0; n < NSTATE; ++n) {
            HIN[(cb2 * NSTATE + n) * DIN + d] = carry[n];
            carry[n] = carry[n] * __expf(An[n] * sd) + HE[(cb2 * NSTATE + n) * DIN + d];
        }
    }
}

// =====================================================================
// Scan pass C: replay chunk with correct h_in, fuse D skip + SiLU(z) gate,
// write y in-place over the dead u-columns of xz (stride 4096).
// =====================================================================
__launch_bounds__(256)
__global__ void scan_passC(const float* __restrict__ dtraw, const float* __restrict__ dtb,
                           const float* __restrict__ u_act, const float* __restrict__ x_dbl,
                           const float* __restrict__ A_log, const float* __restrict__ HIN,
                           const float* __restrict__ Dp, float* __restrict__ xz) {
    const int d = blockIdx.x * 256 + threadIdx.x;
    const int c = blockIdx.y, b = blockIdx.z;
    const float bias = dtb[d];
    const float dskip = Dp[d];
    const size_t cb2 = (size_t)b * NCHUNK + c;
    float An[NSTATE], h[NSTATE];
#pragma unroll
    for (int n = 0; n < NSTATE; ++n) {
        An[n] = -__expf(A_log[d * NSTATE + n]);
        h[n]  = HIN[(cb2 * NSTATE + n) * DIN + d];
    }
    const size_t base = (size_t)b * LSEQ + (size_t)c * CLEN;
    for (int i = 0; i < CLEN; ++i) {
        const size_t row = base + i;
        const float delta = softplusf(dtraw[row * DIN + d] + bias);
        const float u     = u_act[row * DIN + d];
        const float du    = delta * u;
        const float* bc = x_dbl + row * 128 + DTRANK;          // B_t
        const float* cc = x_dbl + row * 128 + DTRANK + NSTATE; // C_t
        float y = 0.0f;
#pragma unroll
        for (int n = 0; n < NSTATE; ++n) {
            h[n] = h[n] * __expf(delta * An[n]) + du * bc[n];
            y += h[n] * cc[n];
        }
        const float z = xz[row * (size_t)(2 * DIN) + DIN + d];
        xz[row * (size_t)(2 * DIN) + d] = (y + u * dskip) * (z * sigmoidf_fast(z));
    }
}

// ---------------- workspace layout (bytes) ----------------
#define OFF_XZ     0ull                              // 8192*4096 f32 = 134217728
#define OFF_UACT   134217728ull                      // 8192*2048 f32 =  67108864
#define OFF_DTRAW  201326592ull                      // 8192*2048 f32 =  67108864
#define OFF_XDBL   268435456ull                      // 8192*128  f32 =   4194304
#define OFF_XWPAD  272629760ull                      // 128*2048  f32 =   1048576
#define OFF_SD     273678336ull                      // 2*32*2048 f32 =    524288
#define OFF_HE     274202624ull                      // 2*32*16*2048  =   8388608
#define OFF_HIN    282591232ull                      // 2*32*16*2048  =   8388608
                                                     // total ~ 291 MB

extern "C" void kernel_launch(void* const* d_in, const int* in_sizes, int n_in,
                              void* d_out, int out_size, void* d_ws, size_t ws_size,
                              hipStream_t stream) {
    const float* x          = (const float*)d_in[0];
    const float* in_proj_w  = (const float*)d_in[1];
    const float* conv_w     = (const float*)d_in[2];
    const float* conv_b     = (const float*)d_in[3];
    const float* x_proj_w   = (const float*)d_in[4];
    const float* dt_proj_w  = (const float*)d_in[5];
    const float* dt_proj_b  = (const float*)d_in[6];
    const float* A_log      = (const float*)d_in[7];
    const float* Dp         = (const float*)d_in[8];
    const float* out_proj_w = (const float*)d_in[9];
    float* out = (float*)d_out;

    char* ws = (char*)d_ws;
    float* xz     = (float*)(ws + OFF_XZ);
    float* u_act  = (float*)(ws + OFF_UACT);
    float* dtraw  = (float*)(ws + OFF_DTRAW);
    float* x_dbl  = (float*)(ws + OFF_XDBL);
    float* xw_pad = (float*)(ws + OFF_XWPAD);
    float* SD     = (float*)(ws + OFF_SD);
    float* HE     = (float*)(ws + OFF_HE);
    float* HIN    = (float*)(ws + OFF_HIN);

    // 0) pad x_proj_w (96 -> 128 rows)
    pad_xproj_w<<<(128 * DIN) / 256, 256, 0, stream>>>(x_proj_w, xw_pad);

    // 1) xz = x @ in_proj_w^T            [8192,1024]x[4096,1024]^T
    gemm_bf16_wmma<<<dim3(2 * DIN / 128, NROWS / 128), 256, 0, stream>>>(
        x, DMODEL, in_proj_w, DMODEL, xz, 2 * DIN, DMODEL);

    // 2) depthwise conv + SiLU -> u_act
    conv_silu_kernel<<<(unsigned)((size_t)NROWS * DIN / 256), 256, 0, stream>>>(
        xz, conv_w, conv_b, u_act);

    // 3) x_dbl = u_act @ x_proj_w_pad^T  [8192,2048]x[128,2048]^T
    gemm_bf16_wmma<<<dim3(1, NROWS / 128), 256, 0, stream>>>(
        u_act, DIN, xw_pad, DIN, x_dbl, 128, DIN);

    // 4) dtraw = dt_lo @ dt_proj_w^T     [8192,64(lda128)]x[2048,64]^T
    gemm_bf16_wmma<<<dim3(DIN / 128, NROWS / 128), 256, 0, stream>>>(
        x_dbl, 128, dt_proj_w, DTRANK, dtraw, DIN, DTRANK);

    // 5-7) chunked selective scan (exact)
    scan_passA<<<dim3(DIN / 256, NCHUNK, BSZ), 256, 0, stream>>>(
        dtraw, dt_proj_b, u_act, x_dbl, A_log, SD, HE);
    scan_passB<<<dim3(DIN / 256, BSZ), 256, 0, stream>>>(SD, HE, A_log, HIN);
    scan_passC<<<dim3(DIN / 256, NCHUNK, BSZ), 256, 0, stream>>>(
        dtraw, dt_proj_b, u_act, x_dbl, A_log, HIN, Dp, xz);

    // 8) out = y @ out_proj_w^T          y lives in xz cols 0..2047 (lda 4096)
    gemm_bf16_wmma<<<dim3(DMODEL / 128, NROWS / 128), 256, 0, stream>>>(
        xz, 2 * DIN, out_proj_w, DIN, out, DMODEL, DIN);
}
